// StochasticTemporalPredictor_45595372814461
// MI455X (gfx1250) — matrix operand order repaired and checked by reading, not compiled, not run on hardware
//
#include <hip/hip_runtime.h>

// ---------------------------------------------------------------------------
// MI455X (gfx1250) implementation: flow-based duration-predictor forward pass.
// All 1x1 convs (the ~290 GFLOP bulk) run on v_wmma_f32_16x16x32_f16 with
// LDS-staged, fragment-swizzled operands and fused bias/LayerNorm/GELU/
// residual/mask epilogues (the GEMMs are HBM-bound at ~64 FLOP/byte, so
// fusion — touching each 64MB tensor once — matters more than peak matmul).
// Round-1 fix: accumulators are now fixed named registers (acc0/acc1) with a
// compile-time tile assignment, eliminating the v_movrels/v_movreld dynamic
// VGPR indexing the previous version generated around every WMMA.
// ---------------------------------------------------------------------------

typedef __attribute__((ext_vector_type(16))) _Float16 v16h;
typedef __attribute__((ext_vector_type(8)))  float    v8f;

constexpr int Bdim = 32;
constexpr int Cdim = 256;
constexpr int Tdim = 2048;
constexpr float LOG2PI_F = 1.8378770664093453f;

// ---------------- device helpers -------------------------------------------
__device__ __forceinline__ float geluf(float x) {
    return 0.5f * x * (1.0f + erff(x * 0.70710678118654752440f));
}
__device__ __forceinline__ float softplusf(float x) {
    return (x > 20.0f) ? x : log1pf(expf(x));
}
__device__ __forceinline__ float sigmoidf(float x) {
    return 1.0f / (1.0f + expf(-x));
}
// log_sigmoid(x) = -softplus(-x)
__device__ __forceinline__ float logsigf(float x) { return -softplusf(-x); }

// block(256)-wide sum; result valid on thread 0 only
__device__ __forceinline__ float blockSum256(float v, float* sh8) {
    #pragma unroll
    for (int o = 16; o >= 1; o >>= 1) v += __shfl_xor(v, o, 32);
    const int lane = threadIdx.x & 31, wv = threadIdx.x >> 5;
    if (lane == 0) sh8[wv] = v;
    __syncthreads();
    float r = 0.0f;
    if (threadIdx.x == 0) {
        #pragma unroll
        for (int i = 0; i < 8; ++i) r += sh8[i];
    }
    return r;
}

// ---------------- fused 1x1 conv (GEMM) on WMMA ----------------------------
// Y[b,o,t] = act( LN?( sum_c W[o,c]*X[b,c,t] + bias[o] ) ) [+res][*mask]
// Block: 256 threads (8 wave32). Weights staged once per WG into LDS in the
// V_WMMA A-fragment swizzle; 16 column tiles (256 t) per WG.
// Wave 'wv' owns M-tiles {wv, wv+8} (MTILES=16) or {wv} if wv<MTILES (=2).
template <int MTILES, bool DO_LN>
__global__ __launch_bounds__(256) void conv1x1_wmma_k(
    const float* __restrict__ Wm, const float* __restrict__ bias,
    const float* __restrict__ X, float* __restrict__ Y,
    const float* __restrict__ res, const float* __restrict__ gamma,
    const float* __restrict__ beta, const float* __restrict__ mask,
    int outc, int mul_mask, int mask_pre_res)
{
    extern __shared__ __align__(128) char smem[];
    _Float16* lw = (_Float16*)smem;            // MTILES*8 frags * 32 lanes * 16 half
    _Float16* lx = lw + MTILES * 8 * 512;      // 8 frags * 32 lanes * 16 half
    __shared__ float red[8][16][2];
    __shared__ float stats[16][2];

    const int tid  = threadIdx.x;
    const int lane = tid & 31;
    const int wv   = tid >> 5;
    const int hh   = lane >> 4;   // half-wave
    const int nn   = lane & 15;

    // ---- stage weights f32->f16 into A-fragment swizzle ----
    // A frag order used here (same as B): lanes0-15 M=0..15; element e -> K=e+16h
    for (int f = wv; f < MTILES * 8; f += 8) {
        const int mt = f >> 3, kt = f & 7;
        const int row = mt * 16 + nn;
        v16h pk;
        if (row < outc) {
            const float* src = Wm + row * Cdim + kt * 32 + 16 * hh;
            #pragma unroll
            for (int e = 0; e < 8; ++e) pk[e] = (_Float16)src[e];
            #pragma unroll
            for (int e = 0; e < 8; ++e) pk[8 + e] = (_Float16)src[8 + e];
        } else {
            #pragma unroll
            for (int e = 0; e < 16; ++e) pk[e] = (_Float16)0.0f;
        }
        *(v16h*)(lw + (f * 32 + lane) * 16) = pk;
    }
    __syncthreads();

    const int blocksPerB = Tdim / 256;
    const int b      = blockIdx.x / blocksPerB;
    const int t0base = (blockIdx.x % blocksPerB) * 256;

    constexpr int NACC = (MTILES > 8) ? 2 : 1;
    const bool act  = (MTILES >= 8) ? true : (wv < MTILES);  // wave-uniform
    const int  mt0  = wv;
    const int  mt1  = wv + 8;

    // per-lane LDS fragment base addresses (constant per wave)
    const _Float16* lxb  = lx + lane * 16;
    const _Float16* lwb0 = lw + (mt0 * 8 * 32 + lane) * 16;
    const _Float16* lwb1 = lw + (mt1 * 8 * 32 + lane) * 16;

    for (int ct = 0; ct < 16; ++ct) {
        const int t0 = t0base + ct * 16;
        // ---- stage X tile into B-fragment swizzle (K=e+16h per lane) ----
        {
            const int xn = tid & 15, q = tid >> 4;   // q*16 = K base
            const float* src = X + ((long)b * Cdim + q * 16) * Tdim + t0 + xn;
            v16h p;
            #pragma unroll
            for (int e = 0; e < 16; ++e) p[e] = (_Float16)src[e * Tdim];
            if (ct + 1 < 16) __builtin_prefetch(src + 16, 0, 0);  // global_prefetch_b8
            __syncthreads();                 // previous tile fully consumed
            *(v16h*)(lx + tid * 16) = p;
        }
        __syncthreads();

        v8f acc0, acc1;
        #pragma unroll
        for (int e = 0; e < 8; ++e) { acc0[e] = 0.0f; acc1[e] = 0.0f; }

        #pragma unroll
        for (int kt = 0; kt < 8; ++kt) {
            v16h bf = *(const v16h*)(lxb + kt * 512);
            if (act) {
                v16h a0 = *(const v16h*)(lwb0 + kt * 512);
                acc0 = __builtin_amdgcn_wmma_f32_16x16x32_f16(
                    false, a0, false, bf, (short)0, acc0, false, false);
                if (NACC == 2) {
                    v16h a1 = *(const v16h*)(lwb1 + kt * 512);
                    acc1 = __builtin_amdgcn_wmma_f32_16x16x32_f16(
                        false, a1, false, bf, (short)0, acc1, false, false);
                }
            }
        }

        const float maskv = mask ? mask[(long)b * Tdim + t0 + nn] : 1.0f;

        // bias (part of LN input)
        if (act) {
            #pragma unroll
            for (int v = 0; v < 8; ++v) {
                const int o0 = mt0 * 16 + v + 8 * hh;
                acc0[v] += (o0 < outc) ? bias[o0] : 0.0f;
            }
            if (NACC == 2) {
                #pragma unroll
                for (int v = 0; v < 8; ++v) {
                    const int o1 = mt1 * 16 + v + 8 * hh;
                    acc1[v] += (o1 < outc) ? bias[o1] : 0.0f;
                }
            }
        }

        if (DO_LN) {   // cross-channel LayerNorm + exact GELU (MTILES==16 only)
            float s1 = 0.0f, s2 = 0.0f;
            #pragma unroll
            for (int v = 0; v < 8; ++v) {
                float a = acc0[v]; s1 += a; s2 += a * a;
                float c2 = acc1[v]; s1 += c2; s2 += c2 * c2;
            }
            s1 += __shfl_xor(s1, 16, 32);
            s2 += __shfl_xor(s2, 16, 32);
            if (lane < 16) { red[wv][nn][0] = s1; red[wv][nn][1] = s2; }
            __syncthreads();
            if (tid < 16) {
                float a = 0.0f, q2 = 0.0f;
                #pragma unroll
                for (int i2 = 0; i2 < 8; ++i2) { a += red[i2][tid][0]; q2 += red[i2][tid][1]; }
                const float mean = a / (float)Cdim;
                const float var  = q2 / (float)Cdim - mean * mean;
                stats[tid][0] = mean;
                stats[tid][1] = rsqrtf(var + 1e-5f);
            }
            __syncthreads();
            const float mean = stats[nn][0], rstd = stats[nn][1];
            #pragma unroll
            for (int v = 0; v < 8; ++v) {
                const int o0 = mt0 * 16 + v + 8 * hh;
                const int o1 = mt1 * 16 + v + 8 * hh;
                acc0[v] = geluf((acc0[v] - mean) * rstd * gamma[o0] + beta[o0]);
                acc1[v] = geluf((acc1[v] - mean) * rstd * gamma[o1] + beta[o1]);
            }
        }

        if (act) {
            #pragma unroll
            for (int v = 0; v < 8; ++v) {
                const int o0 = mt0 * 16 + v + 8 * hh;
                if (o0 < outc) {
                    const long oi = ((long)b * outc + o0) * Tdim + t0 + nn;
                    float a = acc0[v];
                    if (mask_pre_res) a *= maskv;
                    if (res) a += res[oi];
                    if (mul_mask) a *= maskv;
                    Y[oi] = a;
                }
            }
            if (NACC == 2) {
                #pragma unroll
                for (int v = 0; v < 8; ++v) {
                    const int o1 = mt1 * 16 + v + 8 * hh;
                    if (o1 < outc) {
                        const long oi = ((long)b * outc + o1) * Tdim + t0 + nn;
                        float a = acc1[v];
                        if (mask_pre_res) a *= maskv;
                        if (res) a += res[oi];
                        if (mul_mask) a *= maskv;
                        Y[oi] = a;
                    }
                }
            }
        }
    }
}

// ---------------- depthwise dilated conv + LayerNorm + GELU ----------------
__global__ __launch_bounds__(256) void dw_ln_gelu_k(
    const float* __restrict__ xin, const float* __restrict__ mask,
    const float* __restrict__ dww, const float* __restrict__ dwb,
    const float* __restrict__ g1, const float* __restrict__ b1,
    float* __restrict__ yout, int dil)
{
    __shared__ float rs[8], rs2[8], st[2];
    constexpr int TC = 8;
    const int bb = blockIdx.x / (Tdim / TC);
    const int tb = blockIdx.x % (Tdim / TC);
    const int c  = threadIdx.x;
    const float w0 = dww[c * 3 + 0], w1 = dww[c * 3 + 1], w2 = dww[c * 3 + 2];
    const float bz = dwb[c], gg = g1[c], be = b1[c];
    const long base  = ((long)bb * Cdim + c) * Tdim;
    const long mbase = (long)bb * Tdim;

    for (int tt = 0; tt < TC; ++tt) {
        const int t = tb * TC + tt;
        const float xm = (t - dil >= 0)   ? xin[base + t - dil] * mask[mbase + t - dil] : 0.0f;
        const float xc =                    xin[base + t]       * mask[mbase + t];
        const float xp = (t + dil < Tdim) ? xin[base + t + dil] * mask[mbase + t + dil] : 0.0f;
        const float v  = fmaf(w0, xm, fmaf(w1, xc, fmaf(w2, xp, bz)));

        float s1 = v, s2 = v * v;
        #pragma unroll
        for (int o = 16; o >= 1; o >>= 1) { s1 += __shfl_xor(s1, o, 32); s2 += __shfl_xor(s2, o, 32); }
        const int lane = c & 31, wv = c >> 5;
        if (lane == 0) { rs[wv] = s1; rs2[wv] = s2; }
        __syncthreads();
        if (c == 0) {
            float a = 0.0f, q = 0.0f;
            #pragma unroll
            for (int i = 0; i < 8; ++i) { a += rs[i]; q += rs2[i]; }
            const float mean = a / (float)Cdim;
            st[0] = mean;
            st[1] = rsqrtf(q / (float)Cdim - mean * mean + 1e-5f);
        }
        __syncthreads();
        yout[base + t] = geluf((v - st[0]) * st[1] * gg + be);
        __syncthreads();
    }
}

// ---------------- per-channel (C,1) conv: out = w[o]*z + b[o] (+g) ---------
__global__ __launch_bounds__(256) void pre1_k(
    const float* __restrict__ zin, int zstride,
    const float* __restrict__ wv, const float* __restrict__ bv,
    const float* __restrict__ g, float* __restrict__ out)
{
    const int i = blockIdx.x * 256 + threadIdx.x;   // over B*C*T
    const int t = i % Tdim;
    const int o = (i / Tdim) % Cdim;
    const int bb = i / (Tdim * Cdim);
    float v = fmaf(wv[o], zin[(long)bb * zstride + t], bv[o]);
    if (g) v += g[i];
    out[i] = v;
}

// ---------------- rational-quadratic spline + channel flip -----------------
__device__ __forceinline__ void knots10(const float* u, float* cpos, float* wid) {
    float mx = u[0];
    #pragma unroll
    for (int j = 1; j < 10; ++j) mx = fmaxf(mx, u[j]);
    float ex[10], s = 0.0f;
    #pragma unroll
    for (int j = 0; j < 10; ++j) { ex[j] = expf(u[j] - mx); s += ex[j]; }
    const float inv = 1.0f / s;
    float run = 0.0f;
    cpos[0] = -5.0f;
    #pragma unroll
    for (int j = 0; j < 10; ++j) {
        run += 0.001f + 0.99f * ex[j] * inv;    // MIN_BIN + (1-MIN_BIN*nb)*softmax
        cpos[j + 1] = 10.0f * run - 5.0f;
    }
    cpos[10] = 5.0f;
    #pragma unroll
    for (int j = 0; j < 10; ++j) wid[j] = cpos[j + 1] - cpos[j];
}

__global__ __launch_bounds__(256) void rqs_k(
    const float* __restrict__ zin, const float* __restrict__ h29,
    const float* __restrict__ mask, float* __restrict__ zout,
    float* __restrict__ acc, float sign)
{
    __shared__ float sh8[8];
    const int bb = blockIdx.x / (Tdim / 256);
    const int t  = (blockIdx.x % (Tdim / 256)) * 256 + threadIdx.x;
    const float mv = mask[(long)bb * Tdim + t];
    const float z0 = zin[((long)bb * 2 + 0) * Tdim + t];
    const float z1 = zin[((long)bb * 2 + 1) * Tdim + t];

    const float invs = 1.0f / 16.0f;   // 1/sqrt(C)
    float uw[10], uh[10], dk[11];
    #pragma unroll
    for (int j = 0; j < 10; ++j) uw[j] = h29[((long)bb * 29 + j) * Tdim + t] * invs;
    #pragma unroll
    for (int j = 0; j < 10; ++j) uh[j] = h29[((long)bb * 29 + 10 + j) * Tdim + t] * invs;
    dk[0] = 1.0f; dk[10] = 1.0f;       // boundary: MIN_DERIV + softplus(const) = 1
    #pragma unroll
    for (int j = 0; j < 9; ++j)
        dk[j + 1] = 0.001f + softplusf(h29[((long)bb * 29 + 20 + j) * Tdim + t]);

    const bool inside = (z1 >= -5.0f) && (z1 <= 5.0f);
    const float xx = fminf(fmaxf(z1, -5.0f), 5.0f);

    float cw[11], wd[10], ch[11], ht[10];
    knots10(uw, cw, wd);
    knots10(uh, ch, ht);

    int idx = -1;
    #pragma unroll
    for (int k = 0; k < 11; ++k) idx += (xx >= cw[k]) ? 1 : 0;
    idx = (idx < 0) ? 0 : ((idx > 9) ? 9 : idx);

    const float icw = cw[idx], iw = wd[idx], ich = ch[idx], ih = ht[idx];
    const float delta = ih / iw, d0 = dk[idx], d1 = dk[idx + 1];
    const float th = (xx - icw) / iw, th1 = th * (1.0f - th), omt = 1.0f - th;
    const float num = ih * (delta * th * th + d0 * th1);
    const float den = delta + (d0 + d1 - 2.0f * delta) * th1;
    const float outv = ich + num / den;
    const float dnum = delta * delta * (d1 * th * th + 2.0f * delta * th1 + d0 * omt * omt);
    const float lad  = logf(dnum) - 2.0f * logf(den);

    const float z1n  = inside ? outv : z1;
    const float ladv = inside ? lad : 0.0f;

    // write flipped (z = concat(z0, z1n)*mask then z[:, ::-1])
    zout[((long)bb * 2 + 0) * Tdim + t] = z1n * mv;
    zout[((long)bb * 2 + 1) * Tdim + t] = z0 * mv;

    const float s = blockSum256(ladv * mv * sign, sh8);
    if (threadIdx.x == 0) atomicAdd(&acc[bb], s);
}

// ---------------- affine flow: y=(m+e^logs * z)*mask, acc -= logs*mask -----
__global__ __launch_bounds__(256) void affine_k(
    const float* __restrict__ zin, const float* __restrict__ am,
    const float* __restrict__ alogs, const float* __restrict__ mask,
    float* __restrict__ zout, float* __restrict__ acc, int premask)
{
    __shared__ float sh8[8];
    const int i = blockIdx.x * 256 + threadIdx.x;   // over B*2*T
    const int t = i % Tdim;
    const int c = (i / Tdim) & 1;
    const int bb = i / (2 * Tdim);
    const float mv = mask[(long)bb * Tdim + t];
    float z = zin[i];
    if (premask) z *= mv;
    zout[i] = (am[c] + expf(alogs[c]) * z) * mv;
    const float s = blockSum256(-alogs[c] * mv, sh8);
    if (threadIdx.x == 0) atomicAdd(&acc[bb], s);
}

// ---------------- acc += coef*(LOG2PI + val^2)*mask ------------------------
__global__ __launch_bounds__(256) void gauss_k(
    const float* __restrict__ zin, const float* __restrict__ mask,
    float* __restrict__ acc, float coef, int premask)
{
    __shared__ float sh8[8];
    const int i = blockIdx.x * 256 + threadIdx.x;   // over B*2*T
    const int t = i % Tdim;
    const int bb = i / (2 * Tdim);
    const float mv = mask[(long)bb * Tdim + t];
    float v = zin[i];
    if (premask) v *= mv;
    const float s = blockSum256(coef * (LOG2PI_F + v * v) * mv, sh8);
    if (threadIdx.x == 0) atomicAdd(&acc[bb], s);
}

// ---------------- posterior head: u=sig(zu)*m, z0=(w-u)*m, y0=log(...)*m ---
__global__ __launch_bounds__(256) void post_head_k(
    const float* __restrict__ zin, const float* __restrict__ wdur,
    const float* __restrict__ mask, float* __restrict__ zout,
    float* __restrict__ acc)
{
    __shared__ float sh8[8];
    const int bb = blockIdx.x / (Tdim / 256);
    const int t  = (blockIdx.x % (Tdim / 256)) * 256 + threadIdx.x;
    const float mv = mask[(long)bb * Tdim + t];
    const float zu = zin[((long)bb * 2 + 0) * Tdim + t];
    const float z1 = zin[((long)bb * 2 + 1) * Tdim + t];
    const float u  = sigmoidf(zu) * mv;
    const float z0 = (wdur[(long)bb * Tdim + t] - u) * mv;
    const float y0 = logf(fmaxf(z0, 1e-5f)) * mv;
    zout[((long)bb * 2 + 0) * Tdim + t] = y0;
    zout[((long)bb * 2 + 1) * Tdim + t] = z1;
    // out += y0  (from -ldt, ldt += sum(-y0))   out -= (logsig(zu)+logsig(-zu))*m
    const float s = blockSum256(y0 - (logsigf(zu) + logsigf(-zu)) * mv, sh8);
    if (threadIdx.x == 0) atomicAdd(&acc[bb], s);
}

// ===========================================================================
// Host orchestration
// ===========================================================================
struct DDSP { const float *dw_w,*dw_b,*ln1_g,*ln1_b,*pw_w,*pw_b,*ln2_g,*ln2_b; };
struct SPLP { DDSP dds; const float *pre_b,*pre_w,*proj_b,*proj_w; };

// pytree leaf order inside a dds dict (sorted keys):
// dw_b, dw_w, ln1_b, ln1_g, ln2_b, ln2_g, pw_b, pw_w
static DDSP loadDDS(void* const* din, int base) {
    DDSP d;
    d.dw_b  = (const float*)din[base + 0];
    d.dw_w  = (const float*)din[base + 1];
    d.ln1_b = (const float*)din[base + 2];
    d.ln1_g = (const float*)din[base + 3];
    d.ln2_b = (const float*)din[base + 4];
    d.ln2_g = (const float*)din[base + 5];
    d.pw_b  = (const float*)din[base + 6];
    d.pw_w  = (const float*)din[base + 7];
    return d;
}
// spline dict sorted keys: dds(8 leaves), pre_b, pre_w, proj_b, proj_w
static SPLP loadSPL(void* const* din, int base) {
    SPLP s;
    s.dds    = loadDDS(din, base);
    s.pre_b  = (const float*)din[base + 8];
    s.pre_w  = (const float*)din[base + 9];
    s.proj_b = (const float*)din[base + 10];
    s.proj_w = (const float*)din[base + 11];
    return s;
}

static void launch_gemm256(const float* W, const float* b, const float* X, float* Y,
                           const float* res, const float* g, const float* be,
                           const float* mask, int mul_mask, int mask_pre,
                           hipStream_t s)
{
    const size_t SM = (size_t)(16 * 8 * 512 + 4096) * sizeof(_Float16);  // 136 KB
    dim3 grid(Bdim * (Tdim / 256));
    if (g)
        conv1x1_wmma_k<16, true><<<grid, 256, SM, s>>>(W, b, X, Y, res, g, be, mask,
                                                       256, mul_mask, mask_pre);
    else
        conv1x1_wmma_k<16, false><<<grid, 256, SM, s>>>(W, b, X, Y, res, nullptr, nullptr,
                                                        mask, 256, mul_mask, mask_pre);
}

static void launch_gemm29(const float* W, const float* b, const float* X, float* Y,
                          const float* mask, hipStream_t s)
{
    const size_t SM = (size_t)(2 * 8 * 512 + 4096) * sizeof(_Float16);   // 24 KB
    dim3 grid(Bdim * (Tdim / 256));
    conv1x1_wmma_k<2, false><<<grid, 256, SM, s>>>(W, b, X, Y, nullptr, nullptr, nullptr,
                                                   mask, 29, 1, 0);
}

static void run_dds(float* xb, float* tmp, const float* mask, const DDSP& d,
                    hipStream_t s)
{
    int dil = 1;
    for (int i = 0; i < 3; ++i) {
        dw_ln_gelu_k<<<dim3(Bdim * (Tdim / 8)), 256, 0, s>>>(
            xb, mask, d.dw_w + i * Cdim * 3, d.dw_b + i * Cdim,
            d.ln1_g + i * Cdim, d.ln1_b + i * Cdim, tmp, dil);
        // x = x + gelu(ln2(pw(tmp)));  final layer also *mask
        launch_gemm256(d.pw_w + i * Cdim * Cdim, d.pw_b + i * Cdim, tmp, xb,
                       xb, d.ln2_g + i * Cdim, d.ln2_b + i * Cdim, mask,
                       (i == 2) ? 1 : 0, 0, s);
        dil *= 3;
    }
}

static void run_flow(float* zin, float* zout, const float* g, const SPLP& sp,
                     const float* mask, float* bufA, float* tmp, float* h29,
                     float* acc, hipStream_t s)
{
    pre1_k<<<dim3((Bdim * Cdim * Tdim) / 256), 256, 0, s>>>(
        zin, 2 * Tdim, sp.pre_w, sp.pre_b, g, bufA);      // h = pre(z0) + g
    run_dds(bufA, tmp, mask, sp.dds, s);
    launch_gemm29(sp.proj_w, sp.proj_b, bufA, h29, mask, s);
    rqs_k<<<dim3(Bdim * (Tdim / 256)), 256, 0, s>>>(zin, h29, mask, zout, acc, -1.0f);
}

extern "C" void kernel_launch(void* const* d_in, const int* in_sizes, int n_in,
                              void* d_out, int out_size, void* d_ws, size_t ws_size,
                              hipStream_t stream)
{
    (void)in_sizes; (void)n_in; (void)out_size; (void)ws_size;

    const float* x    = (const float*)d_in[0];
    const float* mask = (const float*)d_in[1];
    const float* wdur = (const float*)d_in[2];
    const float* e_q  = (const float*)d_in[3];

    // params pytree leaves (sorted dict keys / list order), starting at 4:
    const float* aff_logs = (const float*)d_in[4];
    const float* aff_m    = (const float*)d_in[5];
    DDSP convs = loadDDS(d_in, 6);
    SPLP flows[4];
    for (int f = 0; f < 4; ++f) flows[f] = loadSPL(d_in, 14 + 12 * f);
    const float* post_aff_logs = (const float*)d_in[62];
    const float* post_aff_m    = (const float*)d_in[63];
    DDSP post_convs = loadDDS(d_in, 64);
    SPLP post_flows[4];
    for (int f = 0; f < 4; ++f) post_flows[f] = loadSPL(d_in, 72 + 12 * f);
    const float* post_pre_b  = (const float*)d_in[120];
    const float* post_pre_w  = (const float*)d_in[121];
    const float* post_proj_b = (const float*)d_in[122];
    const float* post_proj_w = (const float*)d_in[123];
    const float* pre_b       = (const float*)d_in[124];
    const float* pre_w       = (const float*)d_in[125];
    const float* proj_b      = (const float*)d_in[126];
    const float* proj_w      = (const float*)d_in[127];

    // workspace carve-up (floats)
    float* ws = (float*)d_ws;
    const long NBC = (long)Bdim * Cdim * Tdim;       // 16.7M
    float* bufA  = ws;
    float* bufB  = ws + NBC;
    float* xcond = ws + 2 * NBC;                     // persists over 4 flows
    float* gpost = ws + 3 * NBC;                     // persists over 4 post-flows
    float* h29   = ws + 4 * NBC;                     // B*29*T
    float* zA    = h29 + (long)Bdim * 29 * Tdim;     // B*2*T
    float* zB    = zA + (long)Bdim * 2 * Tdim;
    float* acc   = (float*)d_out;                    // per-batch accumulator (B)

    hipMemsetAsync(d_out, 0, Bdim * sizeof(float), stream);

    // ---- main conditioning path: x_cond = proj(dds(pre(x))) * mask ----
    launch_gemm256(pre_w, pre_b, x, bufA, nullptr, nullptr, nullptr, mask, 0, 0, stream);
    run_dds(bufA, bufB, mask, convs, stream);
    launch_gemm256(proj_w, proj_b, bufA, xcond, nullptr, nullptr, nullptr, mask, 1, 0, stream);

    // ---- posterior conditioning: g_post = x_cond + proj(dds(pre(w)))*mask ----
    pre1_k<<<dim3((Bdim * Cdim * Tdim) / 256), 256, 0, stream>>>(
        wdur, Tdim, post_pre_w, post_pre_b, nullptr, bufA);
    run_dds(bufA, bufB, mask, post_convs, stream);
    launch_gemm256(post_proj_w, post_proj_b, bufA, gpost, xcond, nullptr, nullptr,
                   mask, 0, /*mask_pre_res=*/1, stream);

    const int G2 = (Bdim * 2 * Tdim) / 256;

    // logq entropy term: -0.5*(LOG2PI + e^2)*mask, with e = e_q*mask
    gauss_k<<<dim3(G2), 256, 0, stream>>>(e_q, mask, acc, -0.5f, 1);

    // post affine, then 4 post spline flows (each with flip fused)
    affine_k<<<dim3(G2), 256, 0, stream>>>(e_q, post_aff_m, post_aff_logs, mask, zA, acc, 1);
    float* zc = zA; float* za = zB;
    for (int f = 0; f < 4; ++f) {
        run_flow(zc, za, gpost, post_flows[f], mask, bufA, bufB, h29, acc, stream);
        float* t2 = zc; zc = za; za = t2;
    }

    // posterior head: u, z0, y0, log-sigmoid terms
    post_head_k<<<dim3(Bdim * (Tdim / 256)), 256, 0, stream>>>(zc, wdur, mask, za, acc);
    { float* t2 = zc; zc = za; za = t2; }

    // main affine, then 4 spline flows conditioned on x_cond
    affine_k<<<dim3(G2), 256, 0, stream>>>(zc, aff_m, aff_logs, mask, za, acc, 0);
    { float* t2 = zc; zc = za; za = t2; }
    for (int f = 0; f < 4; ++f) {
        run_flow(zc, za, xcond, flows[f], mask, bufA, bufB, h29, acc, stream);
        float* t2 = zc; zc = za; za = t2;
    }

    // final Gaussian NLL term: +0.5*(LOG2PI + z^2)*mask
    gauss_k<<<dim3(G2), 256, 0, stream>>>(zc, mask, acc, 0.5f, 0);
}
